// PointNet2Encoder_75093208203667
// MI455X (gfx1250) — matrix operand order repaired and checked by reading, not compiled, not run on hardware
//
#include <hip/hip_runtime.h>

// ---------------------------------------------------------------------------
// PointNet++ (SSG) encoder for MI455X / gfx1250.
// Dense layers: v_wmma_f32_16x16x32_f16, BN folded into f16 weights on device.
// Big GEMMs: 128x64 block tiles, weight tile double-buffered in LDS via
// GLOBAL_LOAD_ASYNC_TO_LDS_B128 (ASYNCcnt pipeline) when available.
// ---------------------------------------------------------------------------

typedef __attribute__((ext_vector_type(16))) _Float16 v16h;
typedef __attribute__((ext_vector_type(8)))  _Float16 v8h;
typedef __attribute__((ext_vector_type(8)))  float    v8f;
typedef __attribute__((ext_vector_type(4)))  int      v4i;

#define EPS_BN 1e-5f

#if defined(__has_builtin)
#  if __has_builtin(__builtin_amdgcn_global_load_async_to_lds_b128)
#    define HAVE_ASYNC_LDS 1
#  endif
#endif
#ifndef HAVE_ASYNC_LDS
#  define HAVE_ASYNC_LDS 0
#endif

__device__ __forceinline__ void async_copy16(const _Float16* gsrc, _Float16* ldst) {
#if HAVE_ASYNC_LDS
    typedef __attribute__((address_space(1))) v4i* gp_t;   // prints as "int4 __device__*"
    typedef __attribute__((address_space(3))) v4i* lp_t;   // LDS side
    __builtin_amdgcn_global_load_async_to_lds_b128((gp_t)gsrc, (lp_t)ldst, 0, 0);
#else
    uint4 v = *(const uint4*)gsrc;
    *(uint4*)ldst = v;
#endif
}

template <int N>
__device__ __forceinline__ void wait_async_le() {
#if HAVE_ASYNC_LDS
#  if defined(__has_builtin)
#    if __has_builtin(__builtin_amdgcn_s_wait_asynccnt)
    __builtin_amdgcn_s_wait_asynccnt(N);
#    else
    if (N == 0) asm volatile("s_wait_asynccnt 0x0" ::: "memory");
    else        asm volatile("s_wait_asynccnt 0x1" ::: "memory");
#    endif
#  endif
#endif
}

// ---------------------------------------------------------------------------
// Weight prep: Wt[o][i] = W[i][o] * scale[o] (f16, transposed, K padded),
//              bias[o]  = (b-m)*scale + be   (f32).  scale = g*rsqrt(v+eps).
// ---------------------------------------------------------------------------
__global__ void prep_weights_kernel(const float* __restrict__ W,
                                    const float* __restrict__ b,
                                    const float* __restrict__ g,
                                    const float* __restrict__ be,
                                    const float* __restrict__ m,
                                    const float* __restrict__ v,
                                    int ci, int co, int cip, int has_bn,
                                    _Float16* __restrict__ Wt,
                                    float* __restrict__ bias) {
    int gid = blockIdx.x * blockDim.x + threadIdx.x;
    if (gid < co) {
        float sc = has_bn ? g[gid] * rsqrtf(v[gid] + EPS_BN) : 1.0f;
        bias[gid] = has_bn ? (b[gid] - m[gid]) * sc + be[gid] : b[gid];
    }
    if (gid >= co * cip) return;
    int o = gid / cip, i = gid % cip;
    float sc = has_bn ? g[o] * rsqrtf(v[o] + EPS_BN) : 1.0f;
    float w  = (i < ci) ? W[(size_t)i * co + o] * sc : 0.0f;
    Wt[gid] = (_Float16)w;
}

// ---------------------------------------------------------------------------
// Farthest point sampling: one block per batch, dist in global scratch.
// ---------------------------------------------------------------------------
__global__ void fps_kernel(const float* __restrict__ xyz, int N, int npoint,
                           int* __restrict__ out_idx, float* __restrict__ dist) {
    int b = blockIdx.x;
    int t = threadIdx.x;
    const float* px = xyz + (size_t)b * N * 3;
    float* d = dist + (size_t)b * N;
    __shared__ float s_val[256];
    __shared__ int   s_idx[256];
    for (int i = t; i < N; i += blockDim.x) d[i] = 1e10f;
    __syncthreads();
    int sel = 0;
    for (int it = 0; it < npoint; ++it) {
        if (t == 0) out_idx[b * npoint + it] = sel;
        float cx = px[sel * 3 + 0], cy = px[sel * 3 + 1], cz = px[sel * 3 + 2];
        float bestv = -1.0f; int besti = 0;
        for (int i = t; i < N; i += blockDim.x) {
            float dx = px[3 * i] - cx, dy = px[3 * i + 1] - cy, dz = px[3 * i + 2] - cz;
            float dd = dx * dx + dy * dy + dz * dz;
            float od = d[i];
            float nd = od < dd ? od : dd;
            d[i] = nd;
            if (nd > bestv) { bestv = nd; besti = i; }
        }
        s_val[t] = bestv; s_idx[t] = besti;
        __syncthreads();
        for (int s = blockDim.x / 2; s > 0; s >>= 1) {
            if (t < s) {
                bool take = (s_val[t + s] > s_val[t]) ||
                            (s_val[t + s] == s_val[t] && s_idx[t + s] < s_idx[t]);
                if (take) { s_val[t] = s_val[t + s]; s_idx[t] = s_idx[t + s]; }
            }
            __syncthreads();
        }
        sel = s_idx[0];
        __syncthreads();
    }
}

__global__ void gather_centers_kernel(const float* __restrict__ xyz,
                                      const int* __restrict__ idx,
                                      int N, int S, int B, float* __restrict__ out) {
    int gid = blockIdx.x * blockDim.x + threadIdx.x;
    if (gid >= B * S) return;
    int b = gid / S;
    int i = idx[gid];
    const float* p = xyz + ((size_t)b * N + i) * 3;
    out[3 * gid + 0] = p[0];
    out[3 * gid + 1] = p[1];
    out[3 * gid + 2] = p[2];
}

__global__ void ball_query_kernel(const float* __restrict__ xyz,
                                  const float* __restrict__ new_xyz,
                                  int N, int S, int B, int nsample, float r2,
                                  int* __restrict__ out) {
    int gid = blockIdx.x * blockDim.x + threadIdx.x;
    if (gid >= B * S) return;
    int b = gid / S;
    const float* px = xyz + (size_t)b * N * 3;
    float cx = new_xyz[3 * gid + 0], cy = new_xyz[3 * gid + 1], cz = new_xyz[3 * gid + 2];
    int* o = out + (size_t)gid * nsample;
    int cnt = 0, first = N - 1;
    for (int i = 0; i < N && cnt < nsample; ++i) {
        float dx = px[3 * i] - cx, dy = px[3 * i + 1] - cy, dz = px[3 * i + 2] - cz;
        if (dx * dx + dy * dy + dz * dz <= r2) {
            if (cnt == 0) first = i;
            o[cnt++] = i;
        }
    }
    for (; cnt < nsample; ++cnt) o[cnt] = first;
}

__global__ void group_sa_kernel(const float* __restrict__ xyz,
                                const float* __restrict__ new_xyz,
                                const _Float16* __restrict__ feats, // nullable
                                const int* __restrict__ gidx,
                                int N, int S, int B, int ns, int C, int Kp,
                                _Float16* __restrict__ out) {
    int gid = blockIdx.x * blockDim.x + threadIdx.x;
    int total = B * S * ns;
    if (gid >= total) return;
    int bs = gid / ns;
    int b = bs / S;
    int idx = gidx[gid];
    const float* p = xyz + ((size_t)b * N + idx) * 3;
    const float* c = new_xyz + (size_t)bs * 3;
    _Float16* o = out + (size_t)gid * Kp;
    o[0] = (_Float16)(p[0] - c[0]);
    o[1] = (_Float16)(p[1] - c[1]);
    o[2] = (_Float16)(p[2] - c[2]);
    int k = 3;
    if (feats) {
        const _Float16* f = feats + ((size_t)b * N + idx) * C;
        for (int i = 0; i < C; ++i) o[k++] = f[i];
    }
    for (; k < Kp; ++k) o[k] = (_Float16)0.0f;
}

__global__ void group_all_kernel(const float* __restrict__ xyz2,
                                 const _Float16* __restrict__ f2,
                                 int S, int B, int C, int Kp,
                                 _Float16* __restrict__ out) {
    int gid = blockIdx.x * blockDim.x + threadIdx.x;
    if (gid >= B * S) return;
    const float* p = xyz2 + (size_t)gid * 3;
    const _Float16* f = f2 + (size_t)gid * C;
    _Float16* o = out + (size_t)gid * Kp;
    o[0] = (_Float16)p[0]; o[1] = (_Float16)p[1]; o[2] = (_Float16)p[2];
    int k = 3;
    for (int i = 0; i < C; ++i) o[k++] = f[i];
    for (; k < Kp; ++k) o[k] = (_Float16)0.0f;
}

__global__ void maxpool_kernel(const _Float16* __restrict__ in,
                               _Float16* __restrict__ out,
                               int G, int ns, int C) {
    int gid = blockIdx.x * blockDim.x + threadIdx.x;
    if (gid >= G * C) return;
    int g = gid / C, c = gid % C;
    const _Float16* p = in + ((size_t)g * ns) * C + c;
    float mv = -3.0e38f;
    for (int j = 0; j < ns; ++j) {
        float v = (float)p[(size_t)j * C];
        mv = v > mv ? v : mv;
    }
    out[gid] = (_Float16)mv;
}

// ---------------------------------------------------------------------------
// Tiled WMMA GEMM: block = 256 thr / 8 waves, BM=128, BN=64.
// Wave w: rows [bx*128 + w*16, +16), cols [by*64, +64) -> 4 accumulators.
// Weight tile (64 x 32 f16 = 4KB) double-buffered in LDS via async copy.
// Requires M%128==0, N%64==0, Kp%32==0.
// ---------------------------------------------------------------------------
__global__ void wmma_gemm_tiled(const _Float16* __restrict__ A,
                                const _Float16* __restrict__ Wt,
                                const float* __restrict__ bias,
                                _Float16* __restrict__ out,
                                int M, int N, int Kp, int do_relu) {
    __shared__ _Float16 sB[2][64 * 32];   // [buf][n_local*32 + k]
    const int t    = threadIdx.x;
    const int wave = t >> 5;
    const int lane = t & 31;
    const int m    = lane & 15;
    const int half = lane >> 4;
    const int nbase   = blockIdx.y * 64;
    const int rowbase = blockIdx.x * 128 + wave * 16;

    // B-tile copy assignment: thread t moves one 16-byte chunk per stage.
    const int cn = t >> 2;            // 0..63   (n_local)
    const int cc = (t & 3) * 8;       // 0,8,16,24 (k element offset)
    const _Float16* gB = Wt + (size_t)(nbase + cn) * Kp + cc;

    const _Float16* rowA = A + (size_t)(rowbase + m) * Kp;

    const int nk = Kp >> 5;
    async_copy16(gB, &sB[0][cn * 32 + cc]);   // stage k=0

    v8f acc[4] = {v8f{}, v8f{}, v8f{}, v8f{}};
    for (int k = 0; k < nk; ++k) {
        const int buf = k & 1;
        if (k + 1 < nk) {
            async_copy16(gB + (size_t)(k + 1) * 32, &sB[buf ^ 1][cn * 32 + cc]);
            wait_async_le<1>();   // stage k complete (in-order), k+1 in flight
        } else {
            wait_async_le<0>();
        }
        __syncthreads();          // all waves' stage-k data visible

        const int k0 = k << 5;
        v8h alo = *reinterpret_cast<const v8h*>(rowA + k0 + half * 16);
        v8h ahi = *reinterpret_cast<const v8h*>(rowA + k0 + half * 16 + 8);
        v16h a;
#pragma unroll
        for (int i = 0; i < 8; ++i) { a[i] = alo[i]; a[8 + i] = ahi[i]; }

        v16h bfrag[4];
#pragma unroll
        for (int j = 0; j < 4; ++j)
            bfrag[j] = *reinterpret_cast<const v16h*>(&sB[buf][(j * 16 + m) * 32 + half * 16]);

#pragma unroll
        for (int j = 0; j < 4; ++j)
            acc[j] = __builtin_amdgcn_wmma_f32_16x16x32_f16(
                false, a, false, bfrag[j], (short)0, acc[j], false, false);

        __syncthreads();          // reads done before buf is overwritten
    }

#pragma unroll
    for (int j = 0; j < 4; ++j) {
        int ncol = nbase + j * 16 + m;
        float bv = bias[ncol];
#pragma unroll
        for (int r = 0; r < 8; ++r) {
            float v = acc[j][r] + bv;
            if (do_relu) v = v > 0.0f ? v : 0.0f;
            int mrow = rowbase + r + 8 * half;
            out[(size_t)mrow * N + ncol] = (_Float16)v;
        }
    }
}

// ---------------------------------------------------------------------------
// Simple WMMA GEMM for the tiny FC head (M=16): 16x16 tile per wave.
// ---------------------------------------------------------------------------
__global__ void wmma_gemm_bias_act(const _Float16* __restrict__ A,
                                   const _Float16* __restrict__ Wt,
                                   const float* __restrict__ bias,
                                   _Float16* __restrict__ outh,
                                   float* __restrict__ outf,
                                   int M, int N, int Kp,
                                   int do_relu, int store_f32) {
    int wave = threadIdx.x >> 5;
    int lane = threadIdx.x & 31;
    int tileM = blockIdx.x;
    int tileN = blockIdx.y * 4 + wave;    // wave-uniform guard keeps EXEC full
    if (tileN * 16 >= N) return;

    int m    = lane & 15;
    int half = lane >> 4;
    const _Float16* rowA = A  + (size_t)(tileM * 16 + m) * Kp;
    const _Float16* rowB = Wt + (size_t)(tileN * 16 + m) * Kp;

    v8f acc = {};
    for (int k0 = 0; k0 < Kp; k0 += 32) {
        if (k0 + 32 < Kp) {
            __builtin_prefetch(rowA + k0 + 32, 0, 1);
            __builtin_prefetch(rowB + k0 + 32, 0, 1);
        }
        v8h alo = *reinterpret_cast<const v8h*>(rowA + k0 + half * 16);
        v8h ahi = *reinterpret_cast<const v8h*>(rowA + k0 + half * 16 + 8);
        v16h a;
#pragma unroll
        for (int i = 0; i < 8; ++i) { a[i] = alo[i]; a[8 + i] = ahi[i]; }
        v16h bfrag = *reinterpret_cast<const v16h*>(rowB + k0 + half * 16);
        acc = __builtin_amdgcn_wmma_f32_16x16x32_f16(
            false, a, false, bfrag, (short)0, acc, false, false);
    }

    int ncol = tileN * 16 + (lane & 15);
    float bv = bias[ncol];
#pragma unroll
    for (int r = 0; r < 8; ++r) {
        float v = acc[r] + bv;
        if (do_relu) v = v > 0.0f ? v : 0.0f;
        int mrow = tileM * 16 + r + 8 * half;
        if (store_f32) outf[(size_t)mrow * N + ncol] = v;
        else           outh[(size_t)mrow * N + ncol] = (_Float16)v;
    }
}

// ---------------------------------------------------------------------------
static inline size_t align_up(size_t x, size_t a) { return (x + a - 1) & ~(a - 1); }

extern "C" void kernel_launch(void* const* d_in, const int* in_sizes, int n_in,
                              void* d_out, int out_size, void* d_ws, size_t ws_size,
                              hipStream_t stream) {
    (void)in_sizes; (void)n_in; (void)out_size; (void)ws_size;
    const int B = 16, N0 = 4096;
    const int S1 = 512, NS1 = 32;
    const int S2 = 128, NS2 = 64;
    const int M1 = B * S1 * NS1;   // 262144
    const int M2 = B * S2 * NS2;   // 131072
    const int M3 = B * S2;         // 2048
    const int MF = B;              // 16

    const float* xyz = (const float*)d_in[0];

    struct LayerDef { int base, ci, co, cip, bn; };
    const LayerDef L[12] = {
        { 1,    3,   64,   32, 1}, { 7,   64,   64,   64, 1}, {13,   64,  128,   64, 1},
        {19,  131,  128,  160, 1}, {25,  128,  128,  128, 1}, {31,  128,  256,  128, 1},
        {37,  259,  256,  288, 1}, {43,  256,  512,  256, 1}, {49,  512, 1024,  512, 1},
        {55, 1024,  512, 1024, 1}, {61,  512,  256,  512, 1}, {67,  256,   64,  256, 0},
    };

    char* ws = (char*)d_ws;
    size_t cur = 0;
    auto alloc = [&](size_t bytes) { size_t off = cur; cur = align_up(cur + bytes, 256); return off; };

    size_t off_dist  = alloc((size_t)B * N0 * 4);
    size_t off_fidx1 = alloc((size_t)B * S1 * 4);
    size_t off_nx1   = alloc((size_t)B * S1 * 3 * 4);
    size_t off_gidx1 = alloc((size_t)B * S1 * NS1 * 4);
    size_t off_f1    = alloc((size_t)B * S1 * 128 * 2);
    size_t off_fidx2 = alloc((size_t)B * S2 * 4);
    size_t off_nx2   = alloc((size_t)B * S2 * 3 * 4);
    size_t off_gidx2 = alloc((size_t)B * S2 * NS2 * 4);
    size_t off_f2    = alloc((size_t)B * S2 * 256 * 2);
    size_t off_f3    = alloc((size_t)MF * 1024 * 2);
    size_t off_fca1  = alloc((size_t)MF * 512 * 2);
    size_t off_fca2  = alloc((size_t)MF * 256 * 2);

    size_t off_Wt[12], off_bias[12];
    for (int i = 0; i < 12; ++i) {
        off_Wt[i]   = alloc((size_t)L[i].co * L[i].cip * 2);
        off_bias[i] = alloc((size_t)L[i].co * 4);
    }
    size_t off_P0 = alloc((size_t)64 << 20);   // ping
    size_t off_P1 = alloc((size_t)64 << 20);   // pong

    auto f16p = [&](size_t o) { return (_Float16*)(ws + o); };
    auto f32p = [&](size_t o) { return (float*)(ws + o); };
    auto i32p = [&](size_t o) { return (int*)(ws + o); };

    // ---- fold BN into f16 transposed weights ----
    for (int i = 0; i < 12; ++i) {
        const float* W  = (const float*)d_in[L[i].base + 0];
        const float* b  = (const float*)d_in[L[i].base + 1];
        const float* g  = L[i].bn ? (const float*)d_in[L[i].base + 2] : nullptr;
        const float* be = L[i].bn ? (const float*)d_in[L[i].base + 3] : nullptr;
        const float* m  = L[i].bn ? (const float*)d_in[L[i].base + 4] : nullptr;
        const float* v  = L[i].bn ? (const float*)d_in[L[i].base + 5] : nullptr;
        int total = L[i].co * L[i].cip;
        prep_weights_kernel<<<(total + 255) / 256, 256, 0, stream>>>(
            W, b, g, be, m, v, L[i].ci, L[i].co, L[i].cip, L[i].bn,
            f16p(off_Wt[i]), f32p(off_bias[i]));
    }

    // Tiled GEMM for big layers (M%128==0, N%64==0).
    auto gemm_big = [&](const _Float16* A, int li, _Float16* out, int M, int relu) {
        int Nn = L[li].co, Kp = L[li].cip;
        dim3 grd((unsigned)(M / 128), (unsigned)(Nn / 64));
        wmma_gemm_tiled<<<grd, 256, 0, stream>>>(
            A, f16p(off_Wt[li]), f32p(off_bias[li]), out, M, Nn, Kp, relu);
    };
    // Simple GEMM for the FC head (M=16).
    auto gemm_fc = [&](const _Float16* A, int li, void* out, int M, int relu, int f32out) {
        int Nn = L[li].co, Kp = L[li].cip;
        dim3 grd((unsigned)(M / 16), (unsigned)((Nn / 16 + 3) / 4));
        wmma_gemm_bias_act<<<grd, 128, 0, stream>>>(
            A, f16p(off_Wt[li]), f32p(off_bias[li]),
            (_Float16*)out, (float*)out, M, Nn, Kp, relu, f32out);
    };

    // ================= SA1 =================
    fps_kernel<<<B, 256, 0, stream>>>(xyz, N0, S1, i32p(off_fidx1), f32p(off_dist));
    gather_centers_kernel<<<(B * S1 + 255) / 256, 256, 0, stream>>>(
        xyz, i32p(off_fidx1), N0, S1, B, f32p(off_nx1));
    ball_query_kernel<<<(B * S1 + 255) / 256, 256, 0, stream>>>(
        xyz, f32p(off_nx1), N0, S1, B, NS1, 0.2f * 0.2f, i32p(off_gidx1));
    group_sa_kernel<<<(M1 + 255) / 256, 256, 0, stream>>>(
        xyz, f32p(off_nx1), (const _Float16*)nullptr, i32p(off_gidx1),
        N0, S1, B, NS1, 0, 32, f16p(off_P1));
    gemm_big(f16p(off_P1), 0, f16p(off_P0), M1, 1);   // 32 -> 64
    gemm_big(f16p(off_P0), 1, f16p(off_P1), M1, 1);   // 64 -> 64
    gemm_big(f16p(off_P1), 2, f16p(off_P0), M1, 1);   // 64 -> 128
    maxpool_kernel<<<(B * S1 * 128 + 255) / 256, 256, 0, stream>>>(
        f16p(off_P0), f16p(off_f1), B * S1, NS1, 128);

    // ================= SA2 =================
    fps_kernel<<<B, 256, 0, stream>>>(f32p(off_nx1), S1, S2, i32p(off_fidx2), f32p(off_dist));
    gather_centers_kernel<<<(B * S2 + 255) / 256, 256, 0, stream>>>(
        f32p(off_nx1), i32p(off_fidx2), S1, S2, B, f32p(off_nx2));
    ball_query_kernel<<<(B * S2 + 255) / 256, 256, 0, stream>>>(
        f32p(off_nx1), f32p(off_nx2), S1, S2, B, NS2, 0.4f * 0.4f, i32p(off_gidx2));
    group_sa_kernel<<<(M2 + 255) / 256, 256, 0, stream>>>(
        f32p(off_nx1), f32p(off_nx2), f16p(off_f1), i32p(off_gidx2),
        S1, S2, B, NS2, 128, 160, f16p(off_P1));
    gemm_big(f16p(off_P1), 3, f16p(off_P0), M2, 1);   // 160 -> 128
    gemm_big(f16p(off_P0), 4, f16p(off_P1), M2, 1);   // 128 -> 128
    gemm_big(f16p(off_P1), 5, f16p(off_P0), M2, 1);   // 128 -> 256
    maxpool_kernel<<<(B * S2 * 256 + 255) / 256, 256, 0, stream>>>(
        f16p(off_P0), f16p(off_f2), B * S2, NS2, 256);

    // ================= SA3 (group all) =================
    group_all_kernel<<<(M3 + 255) / 256, 256, 0, stream>>>(
        f32p(off_nx2), f16p(off_f2), S2, B, 256, 288, f16p(off_P1));
    gemm_big(f16p(off_P1), 6, f16p(off_P0), M3, 1);   // 288 -> 256
    gemm_big(f16p(off_P0), 7, f16p(off_P1), M3, 1);   // 256 -> 512
    gemm_big(f16p(off_P1), 8, f16p(off_P0), M3, 1);   // 512 -> 1024
    maxpool_kernel<<<(B * 1024 + 255) / 256, 256, 0, stream>>>(
        f16p(off_P0), f16p(off_f3), B, S2, 1024);

    // ================= FC head =================
    gemm_fc(f16p(off_f3),   9, f16p(off_fca1), MF, 1, 0);  // 1024 -> 512
    gemm_fc(f16p(off_fca1), 10, f16p(off_fca2), MF, 1, 0); // 512 -> 256
    gemm_fc(f16p(off_fca2), 11, d_out, MF, 0, 1);          // 256 -> 64, f32 out
}
